// EGraphAttentionLayer_36953898615343
// MI455X (gfx1250) — matrix operand order repaired and checked by reading, not compiled, not run on hardware
//
#include <hip/hip_runtime.h>
#include <hip/hip_fp16.h>
#include <math.h>

#define N_NODES 4096
#define IN_FEAT 256
#define OUT_FEAT 64
#define K_CH 4
#define ALPHA 0.2f
#define NEG_INF -9e15f
#define CCHUNKS 8
#define CHUNK (N_NODES / CCHUNKS)   // 512 rows per partial-softmax chunk
#define SPLITJ 4
#define JTILE (N_NODES / SPLITJ)    // 1024 columns per attmm split

typedef __attribute__((ext_vector_type(16))) _Float16 v16h;
typedef __attribute__((ext_vector_type(8)))  float    v8f;

// ---------------------------------------------------------------------------
// Kernel 1: Wh = h @ W (f32), WhT (f16 transposed, K-contiguous for WMMA B),
//           Wh1[i] = Wh[i,:]·a[:64], Wh2[i] = Wh[i,:]·a[64:]
// ---------------------------------------------------------------------------
__global__ __launch_bounds__(OUT_FEAT)
void wh_kernel(const float* __restrict__ h,
               const float* __restrict__ W,
               const float* __restrict__ a,
               float* __restrict__ Wh,
               _Float16* __restrict__ WhT,
               float* __restrict__ Wh1,
               float* __restrict__ Wh2) {
    __shared__ float sh[IN_FEAT];
    __shared__ float red[OUT_FEAT];
    const int i = blockIdx.x;
    const int f = threadIdx.x;

    for (int c = f; c < IN_FEAT; c += OUT_FEAT)
        sh[c] = h[(size_t)i * IN_FEAT + c];
    __syncthreads();

    float acc = 0.0f;
#pragma unroll 8
    for (int c = 0; c < IN_FEAT; ++c)
        acc = fmaf(sh[c], W[c * OUT_FEAT + f], acc);

    Wh[(size_t)i * OUT_FEAT + f] = acc;
    WhT[(size_t)f * N_NODES + i] = (_Float16)acc;

    red[f] = acc * a[f];
    __syncthreads();
    for (int s = OUT_FEAT / 2; s > 0; s >>= 1) {
        if (f < s) red[f] += red[f + s];
        __syncthreads();
    }
    if (f == 0) Wh1[i] = red[0];
    __syncthreads();

    red[f] = acc * a[OUT_FEAT + f];
    __syncthreads();
    for (int s = OUT_FEAT / 2; s > 0; s >>= 1) {
        if (f < s) red[f] += red[f + s];
        __syncthreads();
    }
    if (f == 0) Wh2[i] = red[0];
}

// ---------------------------------------------------------------------------
// Kernel 2: partial online softmax stats over a 512-row chunk of column j.
// grid (N/256, K, CCHUNKS) -> 4096 waves streaming edge_attr coalesced.
// ---------------------------------------------------------------------------
__global__ __launch_bounds__(256)
void softmax_pass1(const float* __restrict__ edge,
                   const float* __restrict__ Wh1,
                   const float* __restrict__ Wh2,
                   float* __restrict__ pm,
                   float* __restrict__ ps) {
    __shared__ float s1[CHUNK];
    const int k  = blockIdx.y;
    const int c  = blockIdx.z;
    const int j  = blockIdx.x * 256 + threadIdx.x;
    const int i0 = c * CHUNK;

    for (int idx = threadIdx.x; idx < CHUNK; idx += 256)
        s1[idx] = Wh1[i0 + idx];
    __syncthreads();

    const float wh2j = Wh2[j];
    const float* ecol = edge + (size_t)k * N_NODES * N_NODES
                             + (size_t)i0 * N_NODES + j;

    float m = -INFINITY;
    float s = 0.0f;
    for (int i = 0; i < CHUNK; ++i) {
        if (i + 8 < CHUNK)
            __builtin_prefetch(ecol + (size_t)(i + 8) * N_NODES, 0, 1);
        const float ev  = ecol[(size_t)i * N_NODES];
        const float rel = s1[i] + wh2j;
        const float lr  = rel > 0.0f ? rel : ALPHA * rel;
        const float v   = ev > 0.0f ? lr * ev : NEG_INF;
        if (v > m) {
            s = s * expf(m - v) + 1.0f;   // first iter: 0*exp(-inf)+1 = 1
            m = v;
        } else {
            s += expf(v - m);
        }
    }
    const size_t o = (size_t)c * K_CH * N_NODES + (size_t)k * N_NODES + j;
    pm[o] = m;
    ps[o] = s;
}

// ---------------------------------------------------------------------------
// Kernel 3: merge chunk partials -> final per-column max M and 1/sum.
// Exactly reproduces reference semantics (empty column -> M=NEG_INF, S=N).
// ---------------------------------------------------------------------------
__global__ __launch_bounds__(256)
void softmax_combine(const float* __restrict__ pm,
                     const float* __restrict__ ps,
                     float* __restrict__ Mf,
                     float* __restrict__ Inv) {
    const int t = blockIdx.x * 256 + threadIdx.x;   // over K*N columns
    float M = -INFINITY;
#pragma unroll
    for (int c = 0; c < CCHUNKS; ++c)
        M = fmaxf(M, pm[(size_t)c * K_CH * N_NODES + t]);
    float S = 0.0f;
#pragma unroll
    for (int c = 0; c < CCHUNKS; ++c) {
        const size_t o = (size_t)c * K_CH * N_NODES + t;
        S += ps[o] * expf(pm[o] - M);
    }
    Mf[t]  = M;
    Inv[t] = 1.0f / S;
}

// ---------------------------------------------------------------------------
// Kernel 4 (fused, split-K over j): recompute e from edge_attr, normalize
// with (Mf, Inv), write att (f32 output), feed f16 A into
// v_wmma_f32_16x16x32_f16.  Partial 16x64 accumulators go to workspace.
// grid = SPLITJ * K * 256 tiles, one wave each -> 4096 waves.
// ---------------------------------------------------------------------------
__global__ __launch_bounds__(32)
void attmm_fused(const float* __restrict__ edge,
                 const _Float16* __restrict__ WhT,
                 const float* __restrict__ Wh1,
                 const float* __restrict__ Wh2,
                 const float* __restrict__ Mf,
                 const float* __restrict__ Inv,
                 float* __restrict__ att,
                 float* __restrict__ pacc) {
    const int bid  = blockIdx.x;
    const int sp   = bid >> 10;            // 0..SPLITJ-1
    const int rem  = bid & 1023;
    const int k    = rem >> 8;             // 0..3
    const int tile = rem & 255;
    const int i0   = tile << 4;            // row tile base
    const int lane = threadIdx.x;
    const int hl   = lane & 15;
    const int hi   = lane >> 4;            // 0 or 1

    const int jbeg = sp * JTILE;
    const int jend = jbeg + JTILE;

    const int row  = i0 + hl;
    const float wh1 = Wh1[row];
    const float* erow = edge + (size_t)k * N_NODES * N_NODES
                             + (size_t)row * N_NODES;
    float*       arow = att  + (size_t)k * N_NODES * N_NODES
                             + (size_t)row * N_NODES;
    const float* Mrow = Mf  + (size_t)k * N_NODES;
    const float* Irow = Inv + (size_t)k * N_NODES;

    v8f acc[4] = {};   // 4 N-tiles of 16 -> OUT_FEAT = 64

    for (int j0 = jbeg; j0 < jend; j0 += 32) {
        if (j0 + 256 < jend)
            __builtin_prefetch(erow + j0 + 256 + hi * 8, 0, 1);

        // A tile, K-contiguous to match B:
        // lane<16: halfs[0..15]=K{0..15}
        // lane>=16: halfs[0..15]=K{16..31}
        v16h av;
#pragma unroll
        for (int half = 0; half < 2; ++half) {
            const int base = j0 + hi * 16 + half * 8;
#pragma unroll
            for (int q = 0; q < 2; ++q) {
                const int b = base + q * 4;
                const float4 e4 = *(const float4*)(erow + b);
                const float4 w4 = *(const float4*)(Wh2  + b);
                const float4 m4 = *(const float4*)(Mrow + b);
                const float4 i4 = *(const float4*)(Irow + b);

                float4 o4;
                {
                    float r = wh1 + w4.x, lr = r > 0.0f ? r : ALPHA * r;
                    float v = e4.x > 0.0f ? lr * e4.x : NEG_INF;
                    o4.x = expf(v - m4.x) * i4.x;
                }
                {
                    float r = wh1 + w4.y, lr = r > 0.0f ? r : ALPHA * r;
                    float v = e4.y > 0.0f ? lr * e4.y : NEG_INF;
                    o4.y = expf(v - m4.y) * i4.y;
                }
                {
                    float r = wh1 + w4.z, lr = r > 0.0f ? r : ALPHA * r;
                    float v = e4.z > 0.0f ? lr * e4.z : NEG_INF;
                    o4.z = expf(v - m4.z) * i4.z;
                }
                {
                    float r = wh1 + w4.w, lr = r > 0.0f ? r : ALPHA * r;
                    float v = e4.w > 0.0f ? lr * e4.w : NEG_INF;
                    o4.w = expf(v - m4.w) * i4.w;
                }
                *(float4*)(arow + b) = o4;   // att output (f32)

                const int ho = half * 8 + q * 4;
                av[ho + 0] = (_Float16)o4.x;
                av[ho + 1] = (_Float16)o4.y;
                av[ho + 2] = (_Float16)o4.z;
                av[ho + 3] = (_Float16)o4.w;
            }
        }

        // B tiles: 32x16 f16; lane<16 holds K 0..15, lane>=16 holds K 16..31,
        // column N = hl; K-contiguous thanks to transposed WhT.
        v16h bv[4];
#pragma unroll
        for (int t = 0; t < 4; ++t)
            bv[t] = *(const v16h*)(WhT + (size_t)(t * 16 + hl) * N_NODES
                                       + j0 + hi * 16);
#pragma unroll
        for (int t = 0; t < 4; ++t)
            acc[t] = __builtin_amdgcn_wmma_f32_16x16x32_f16(
                false, av, false, bv[t], (short)0, acc[t], false, false);
    }

    // store partial accumulators: [sp][k][tile] block of 16x64, per-lane 32f
    float* pdst = pacc
        + ((size_t)(sp * K_CH + k) * 256 + tile) * (16 * OUT_FEAT)
        + (size_t)lane * 32;
#pragma unroll
    for (int t = 0; t < 4; ++t)
        *(v8f*)(pdst + t * 8) = acc[t];
}

// ---------------------------------------------------------------------------
// Kernel 5: sum the SPLITJ partials, apply ELU, write concat output
// out[i, k*64+f].  idx enumerates (k, i, f).
// ---------------------------------------------------------------------------
__global__ __launch_bounds__(256)
void reduce_elu(const float* __restrict__ pacc,
                float* __restrict__ out) {
    const int idx  = blockIdx.x * 256 + threadIdx.x;   // over K*N*64
    const int f    = idx & 63;
    const int rest = idx >> 6;
    const int i    = rest & (N_NODES - 1);
    const int k    = rest >> 12;

    // invert the per-lane accumulator layout used by attmm_fused
    const int tile = i >> 4;
    const int r    = i & 7;
    const int hi   = (i >> 3) & 1;
    const int lane = hi * 16 + (f & 15);
    const int t    = f >> 4;
    const size_t base = ((size_t)k * 256 + tile) * (16 * OUT_FEAT)
                      + (size_t)lane * 32 + t * 8 + r;
    const size_t spstride = (size_t)K_CH * N_NODES * OUT_FEAT;

    float s = 0.0f;
#pragma unroll
    for (int sp = 0; sp < SPLITJ; ++sp)
        s += pacc[(size_t)sp * spstride + base];

    const float e = s > 0.0f ? s : expm1f(s);
    out[(size_t)i * (K_CH * OUT_FEAT) + k * OUT_FEAT + f] = e;
}

// ---------------------------------------------------------------------------
extern "C" void kernel_launch(void* const* d_in, const int* in_sizes, int n_in,
                              void* d_out, int out_size, void* d_ws, size_t ws_size,
                              hipStream_t stream) {
    const float* h    = (const float*)d_in[0];
    const float* edge = (const float*)d_in[1];
    const float* W    = (const float*)d_in[2];
    const float* a    = (const float*)d_in[3];

    float* out_elu = (float*)d_out;                                  // [N, K*64]
    float* out_att = out_elu + (size_t)N_NODES * K_CH * OUT_FEAT;    // [K, N, N]

    // workspace layout (sequential carve-out, ~19 MB total)
    char* ws = (char*)d_ws;
    float*    Wh   = (float*)ws;              ws += (size_t)N_NODES * OUT_FEAT * 4;
    _Float16* WhT  = (_Float16*)ws;           ws += (size_t)N_NODES * OUT_FEAT * 2;
    float*    Wh1  = (float*)ws;              ws += (size_t)N_NODES * 4;
    float*    Wh2  = (float*)ws;              ws += (size_t)N_NODES * 4;
    float*    Mf   = (float*)ws;              ws += (size_t)K_CH * N_NODES * 4;
    float*    Inv  = (float*)ws;              ws += (size_t)K_CH * N_NODES * 4;
    float*    pm   = (float*)ws;              ws += (size_t)CCHUNKS * K_CH * N_NODES * 4;
    float*    ps   = (float*)ws;              ws += (size_t)CCHUNKS * K_CH * N_NODES * 4;
    float*    pacc = (float*)ws;              // SPLITJ*K*N*64*4 = 16 MB

    wh_kernel<<<N_NODES, OUT_FEAT, 0, stream>>>(h, W, a, Wh, WhT, Wh1, Wh2);

    softmax_pass1<<<dim3(N_NODES / 256, K_CH, CCHUNKS), 256, 0, stream>>>(
        edge, Wh1, Wh2, pm, ps);

    softmax_combine<<<(K_CH * N_NODES) / 256, 256, 0, stream>>>(pm, ps, Mf, Inv);

    attmm_fused<<<SPLITJ * K_CH * (N_NODES / 16), 32, 0, stream>>>(
        edge, WhT, Wh1, Wh2, Mf, Inv, out_att, pacc);

    reduce_elu<<<(K_CH * N_NODES * OUT_FEAT) / 256, 256, 0, stream>>>(
        pacc, out_elu);
}
